// MyModel_32701880992340
// MI455X (gfx1250) — matrix operand order repaired and checked by reading, not compiled, not run on hardware
//
#include <hip/hip_runtime.h>
#include <hip/hip_bf16.h>
#include <math.h>

// ---- shapes ----
#define HID   768
#define BATCH 64
#define SEQ   229
#define TWROWS 12544   // 64 * 196 token-embedding rows

typedef __attribute__((ext_vector_type(16))) _Float16 v16h;
typedef __attribute__((ext_vector_type(8)))  _Float16 v8h;
typedef __attribute__((ext_vector_type(8)))  float    v8f;

// Repack W (K x 768, row-major f32) into f16 layout Bp[k/32][n][k%32] so that a
// lane's 16-element WMMA B fragment (K-contiguous per column) is 2x16B loads.
__global__ __launch_bounds__(256) void pack_w(const float* __restrict__ W,
                                              _Float16* __restrict__ Bp, int K) {
  int idx = blockIdx.x * blockDim.x + threadIdx.x;
  if (idx >= K * HID) return;
  int k = idx / HID, n = idx - k * HID;
  Bp[(((size_t)(k >> 5)) * HID + n) * 32 + (k & 31)] = (_Float16)W[idx];
}

// Gather word_emb rows for every token slot -> f16 A buffer; record dst/seg.
__global__ __launch_bounds__(256) void gather_tokens(
    const float* __restrict__ word_emb,
    const int* __restrict__ cls_id, const int* __restrict__ sep_id,
    const int* __restrict__ user_info, const int* __restrict__ author_info,
    const int* __restrict__ video_info,
    _Float16* __restrict__ A, int* __restrict__ dstIdx, int* __restrict__ segIdx) {
  int row = blockIdx.x;                 // 0..12543
  int b = row / 196, j = row - b * 196;
  int tok, pos, seg;
  if      (j == 0)  { tok = cls_id[b];                  pos = 0;             seg = 0; }
  else if (j <= 32) { tok = user_info[b*32 + j - 1];    pos = 1 + j;         seg = 0; }
  else if (j == 33) { tok = sep_id[b];                  pos = 34;            seg = 0; }
  else if (j <= 65) { tok = author_info[b*32 + j - 34]; pos = 36 + (j - 34); seg = 1; }
  else if (j == 66) { tok = sep_id[b];                  pos = 68;            seg = 1; }
  else if (j <= 194){ tok = video_info[b*128 + j - 67]; pos = 69 + (j - 67); seg = 2; }
  else              { tok = sep_id[b];                  pos = 228;           seg = 2; }
  if (threadIdx.x == 0) { dstIdx[row] = b * SEQ + pos; segIdx[row] = seg; }
  const float* src = word_emb + (size_t)tok * HID;
  _Float16* dst = A + (size_t)row * HID;
  for (int h = threadIdx.x; h < HID; h += blockDim.x) dst[h] = (_Float16)src[h];
}

__global__ __launch_bounds__(256) void cvt_f32_f16(const float* __restrict__ src,
                                                   _Float16* __restrict__ dst, int n) {
  for (int i = blockIdx.x * blockDim.x + threadIdx.x; i < n; i += gridDim.x * blockDim.x)
    dst[i] = (_Float16)src[i];
}

// All non-GEMM rows + attention mask. One block per batch element.
__global__ __launch_bounds__(256) void misc_k(
    const float* __restrict__ user_table, const float* __restrict__ author_table,
    const float* __restrict__ hot_t, const float* __restrict__ ent_t,
    const float* __restrict__ soc_t, const float* __restrict__ chal_t,
    const float* __restrict__ auth_t, const float* __restrict__ type_table,
    const float* __restrict__ Wl, const float* __restrict__ bl,
    const float* __restrict__ ln_g, const float* __restrict__ ln_b,
    const float* __restrict__ live_score,
    const int* __restrict__ user_id, const int* __restrict__ author_id,
    const int* __restrict__ vhot, const int* __restrict__ vent,
    const int* __restrict__ vsoc, const int* __restrict__ vchal,
    const int* __restrict__ vauth,
    const int* __restrict__ ui_mask, const int* __restrict__ ai_mask,
    const int* __restrict__ vi_mask, const int* __restrict__ act_mask,
    const int* __restrict__ mus_mask,
    float* __restrict__ emb, float* __restrict__ attn) {
  int b = blockIdx.x, tid = threadIdx.x;
  __shared__ float xs[HID];
  __shared__ float red[256];
  const float* t0 = type_table;
  const float* t1 = type_table + HID;
  const float* t2 = type_table + 2 * HID;
  const float* urow = user_table   + (size_t)user_id[b]   * HID;
  const float* arow = author_table + (size_t)author_id[b] * HID;
  const float* h1 = hot_t  + vhot[b]  * HID;
  const float* h2 = ent_t  + vent[b]  * HID;
  const float* h3 = soc_t  + vsoc[b]  * HID;
  const float* h4 = chal_t + vchal[b] * HID;
  const float* h5 = auth_t + vauth[b] * HID;
  float* eb = emb + (size_t)b * SEQ * HID;
  for (int h = tid; h < HID; h += 256) {
    eb[(size_t)1   * HID + h] = urow[h] + t0[h];
    eb[(size_t)35  * HID + h] = arow[h] + t1[h];
    eb[(size_t)198 * HID + h] = h1[h] + t2[h];
    eb[(size_t)199 * HID + h] = h2[h] + t2[h];
    eb[(size_t)200 * HID + h] = h3[h] + t2[h];
    eb[(size_t)201 * HID + h] = h4[h] + t2[h];
    eb[(size_t)202 * HID + h] = h5[h] + t2[h];
  }
  // live row: tanh(score*Wl + bl) -> LayerNorm -> *g + b -> + type2
  float sc = live_score[b];
  float s = 0.f;
  for (int h = tid; h < HID; h += 256) { float x = tanhf(sc * Wl[h] + bl[h]); xs[h] = x; s += x; }
  red[tid] = s; __syncthreads();
  for (int o = 128; o; o >>= 1) { if (tid < o) red[tid] += red[tid + o]; __syncthreads(); }
  float mean = red[0] * (1.0f / HID);
  __syncthreads();
  float s2 = 0.f;
  for (int h = tid; h < HID; h += 256) { float d = xs[h] - mean; s2 += d * d; }
  red[tid] = s2; __syncthreads();
  for (int o = 128; o; o >>= 1) { if (tid < o) red[tid] += red[tid + o]; __syncthreads(); }
  float inv = rsqrtf(red[0] * (1.0f / HID) + 1e-5f);
  for (int h = tid; h < HID; h += 256)
    eb[(size_t)197 * HID + h] = (xs[h] - mean) * inv * ln_g[h] + ln_b[h] + t2[h];
  // attention mask
  float* at = attn + b * SEQ;
  for (int p = tid; p < SEQ; p += 256) {
    float v;
    if      (p < 2)    v = 1.f;
    else if (p < 34)   v = (float)ui_mask[b * 32 + p - 2];
    else if (p < 36)   v = 1.f;
    else if (p < 68)   v = (float)ai_mask[b * 32 + p - 36];
    else if (p == 68)  v = 1.f;
    else if (p < 197)  v = (float)vi_mask[b * 128 + p - 69];
    else if (p < 203)  v = 1.f;
    else if (p < 219)  v = (float)act_mask[b * 16 + p - 203];
    else if (p == 219) v = 1.f;
    else if (p < 228)  v = (float)mus_mask[b * 8 + p - 220];
    else               v = 1.f;
    at[p] = v;
  }
}

// One wave computes a 16x64 C strip = 4 16x16 WMMA tiles sharing one A
// fragment (4x A reuse in registers). 4 waves/block, grid=(Mrows/16, 3):
// ngroup = blockIdx.y*4+wave in [0,12) covers N=768 exactly -> EXEC all-1s.
__global__ __launch_bounds__(128) void gemm_wmma(
    const _Float16* __restrict__ A,   // Mrows x K, row-major
    const _Float16* __restrict__ Bp,  // [K/32][768][32] packed
    const float* __restrict__ bias, const float* __restrict__ type_table,
    const int* __restrict__ dstIdx, const int* __restrict__ segIdx,
    int segConst, int tokPerB, int posBase, int Mrows, int K,
    float* __restrict__ emb) {
  const int wave = threadIdx.x >> 5;
  const int lane = threadIdx.x & 31;
  const int ng   = blockIdx.y * 4 + wave;     // N group (64 columns each)
  const int m0   = blockIdx.x << 4;
  const int half = lane >> 4;                 // 0 | 1
  const int l16  = lane & 15;
  const int hk   = half << 3;                 // A K sub-offset: 0 | 8
  const int kb   = half << 3;                 // B K sub-offset: 0 | 8
  const int n0   = (ng << 6) + l16;           // first of 4 columns (stride 16)

  v8f acc[4] = {};
  const int KT = K >> 5;
  const _Float16* Arow = A + (size_t)(m0 + l16) * K;
  const _Float16* B0   = Bp + (size_t)n0 * 32 + kb;
  for (int kt = 0; kt < KT; ++kt) {
    // prefetch A fragments a few k-steps ahead (global_prefetch_b8)
    __builtin_prefetch((const void*)(Arow + ((kt + 4) << 5) + hk), 0, 0);
    // A fragment (ISA 16-bit A 16x32): a[0..7]=K(k0+hk..+7), a[8..15]=K(k0+16+hk..+7)
    const v8h* ap = (const v8h*)(Arow + (kt << 5) + hk);
    v8h a0 = ap[0];
    v8h a1 = ap[2];
    v16h a;
#pragma unroll
    for (int j = 0; j < 8; ++j) { a[j] = a0[j]; a[8 + j] = a1[j]; }
    const _Float16* Bk = B0 + (size_t)kt * HID * 32;
#pragma unroll
    for (int t = 0; t < 4; ++t) {
      // B fragment: b[0..7]=B[k0+kb..+7][n0+16t], b[8..15]=B[k0+16+kb..+7][n0+16t]
      const v8h* bp = (const v8h*)(Bk + t * 16 * 32);
      v8h b0 = bp[0];
      v8h b1 = bp[2];
      v16h bf;
#pragma unroll
      for (int j = 0; j < 8; ++j) { bf[j] = b0[j]; bf[8 + j] = b1[j]; }
      acc[t] = __builtin_amdgcn_wmma_f32_16x16x32_f16(
          false, a, false, bf, (short)0, acc[t], false, false);
    }
  }
#pragma unroll
  for (int r = 0; r < 8; ++r) {
    int mm = m0 + r + (half << 3);            // C layout: lanes 16-31 hold M=8..15
    if (mm >= Mrows) continue;
    int dst, seg;
    if (dstIdx) { dst = dstIdx[mm]; seg = segIdx[mm]; }
    else {
      int bb = mm / tokPerB;
      dst = bb * SEQ + posBase + (mm - bb * tokPerB);
      seg = segConst;
    }
    float* out = emb + (size_t)dst * HID;
    const float* ty = type_table + seg * HID;
#pragma unroll
    for (int t = 0; t < 4; ++t) {
      int n = n0 + (t << 4);
      out[n] = acc[t][r] + bias[n] + ty[n];
    }
  }
}

extern "C" void kernel_launch(void* const* d_in, const int* in_sizes, int n_in,
                              void* d_out, int out_size, void* d_ws, size_t ws_size,
                              hipStream_t stream) {
  (void)in_sizes; (void)n_in; (void)out_size; (void)ws_size;
  const float* word_emb    = (const float*)d_in[0];
  const float* Wt          = (const float*)d_in[1];
  const float* bt          = (const float*)d_in[2];
  const float* user_table  = (const float*)d_in[3];
  const float* author_table= (const float*)d_in[4];
  const float* hot_t       = (const float*)d_in[5];
  const float* ent_t       = (const float*)d_in[6];
  const float* soc_t       = (const float*)d_in[7];
  const float* chal_t      = (const float*)d_in[8];
  const float* auth_t      = (const float*)d_in[9];
  const float* type_table  = (const float*)d_in[10];
  const float* Wa          = (const float*)d_in[11];
  const float* ba          = (const float*)d_in[12];
  const float* Ws          = (const float*)d_in[13];
  const float* bs          = (const float*)d_in[14];
  const float* Wm          = (const float*)d_in[15];
  const float* bm          = (const float*)d_in[16];
  const float* Wl          = (const float*)d_in[17];
  const float* bl          = (const float*)d_in[18];
  const float* ln_g        = (const float*)d_in[19];
  const float* ln_b        = (const float*)d_in[20];
  const float* live        = (const float*)d_in[21];
  const float* act_f       = (const float*)d_in[22];
  const float* sem_f       = (const float*)d_in[23];
  const float* mus_f       = (const float*)d_in[24];
  const int* user_id     = (const int*)d_in[25];
  const int* user_info   = (const int*)d_in[26];
  const int* ui_mask     = (const int*)d_in[27];
  const int* author_id   = (const int*)d_in[28];
  const int* author_info = (const int*)d_in[29];
  const int* ai_mask     = (const int*)d_in[30];
  const int* vhot        = (const int*)d_in[31];
  const int* vent        = (const int*)d_in[32];
  const int* vsoc        = (const int*)d_in[33];
  const int* vchal       = (const int*)d_in[34];
  const int* vauth       = (const int*)d_in[35];
  const int* video_info  = (const int*)d_in[36];
  const int* act_mask    = (const int*)d_in[37];
  const int* mus_mask    = (const int*)d_in[38];
  const int* vi_mask     = (const int*)d_in[39];
  const int* cls_id      = (const int*)d_in[40];
  const int* sep_id      = (const int*)d_in[41];

  float* emb  = (float*)d_out;
  float* attn = emb + (size_t)BATCH * SEQ * HID;

  // carve workspace (~31 MB total)
  char* p = (char*)d_ws;
  auto carve = [&](size_t bytes) -> void* {
    void* r = (void*)p; p += (bytes + 255) & ~(size_t)255; return r;
  };
  _Float16* Bp_t = (_Float16*)carve((size_t)768  * HID * 2);
  _Float16* Bp_a = (_Float16*)carve((size_t)2048 * HID * 2);
  _Float16* Bp_s = (_Float16*)carve((size_t)512  * HID * 2);
  _Float16* Bp_m = (_Float16*)carve((size_t)768  * HID * 2);
  _Float16* A1   = (_Float16*)carve((size_t)TWROWS * HID * 2);
  _Float16* A2   = (_Float16*)carve((size_t)1024 * 2048 * 2);
  _Float16* A3   = (_Float16*)carve((size_t)64 * 512 * 2);
  _Float16* A4   = (_Float16*)carve((size_t)512 * HID * 2);
  int* dstIdx    = (int*)carve((size_t)TWROWS * 4);
  int* segIdx    = (int*)carve((size_t)TWROWS * 4);

  // stage operands
  pack_w<<<(768  * HID + 255) / 256, 256, 0, stream>>>(Wt, Bp_t, 768);
  pack_w<<<(2048 * HID + 255) / 256, 256, 0, stream>>>(Wa, Bp_a, 2048);
  pack_w<<<(512  * HID + 255) / 256, 256, 0, stream>>>(Ws, Bp_s, 512);
  pack_w<<<(768  * HID + 255) / 256, 256, 0, stream>>>(Wm, Bp_m, 768);
  gather_tokens<<<TWROWS, 256, 0, stream>>>(word_emb, cls_id, sep_id,
                                            user_info, author_info, video_info,
                                            A1, dstIdx, segIdx);
  cvt_f32_f16<<<2048, 256, 0, stream>>>(act_f, A2, 64 * 16 * 2048);
  cvt_f32_f16<<<128,  256, 0, stream>>>(sem_f, A3, 64 * 512);
  cvt_f32_f16<<<512,  256, 0, stream>>>(mus_f, A4, 64 * 8 * HID);

  // non-GEMM rows + attention mask
  misc_k<<<BATCH, 256, 0, stream>>>(user_table, author_table,
                                    hot_t, ent_t, soc_t, chal_t, auth_t,
                                    type_table, Wl, bl, ln_g, ln_b, live,
                                    user_id, author_id,
                                    vhot, vent, vsoc, vchal, vauth,
                                    ui_mask, ai_mask, vi_mask, act_mask, mus_mask,
                                    emb, attn);

  dim3 blk(128);
  // token GEMM: (12544 x 768) @ (768 x 768), scatter via dstIdx/segIdx
  gemm_wmma<<<dim3(TWROWS / 16, 3), blk, 0, stream>>>(
      A1, Bp_t, bt, type_table, dstIdx, segIdx, 0, 1, 0, TWROWS, 768, emb);
  // action: (1024 x 2048) @ (2048 x 768) -> positions 203..218
  gemm_wmma<<<dim3(1024 / 16, 3), blk, 0, stream>>>(
      A2, Bp_a, ba, type_table, nullptr, nullptr, 2, 16, 203, 1024, 2048, emb);
  // semantic: (64 x 512) @ (512 x 768) -> position 219
  gemm_wmma<<<dim3(64 / 16, 3), blk, 0, stream>>>(
      A3, Bp_s, bs, type_table, nullptr, nullptr, 2, 1, 219, 64, 512, emb);
  // music: (512 x 768) @ (768 x 768) -> positions 220..227
  gemm_wmma<<<dim3(512 / 16, 3), blk, 0, stream>>>(
      A4, Bp_m, bm, type_table, nullptr, nullptr, 2, 8, 220, 512, 768, emb);
}
